// SimpleGNNEncoder_850403524857
// MI455X (gfx1250) — matrix operand order repaired and checked by reading, not compiled, not run on hardware
//
#include <hip/hip_runtime.h>
#include <hip/hip_bf16.h>

// ---------------- problem constants (match reference) ----------------
#define N_NODES   20000
#define M_PAD     20480              // padded row count: 1280 tiles = 40 panels x 32
#define E_EDGES   320000
#define E_TOT     (E_EDGES + N_NODES)   // with self loops
#define IN_DIM    64
#define HID       128
#define HEADS     4
#define LAYERS    3
#define N_GRAPHS  64
#define NEG_SLOPE 0.2f
#define LN_EPS    1e-5f

typedef __attribute__((ext_vector_type(16))) _Float16 v16h;
typedef __attribute__((ext_vector_type(8)))  _Float16 v8h;
typedef __attribute__((ext_vector_type(8)))  float    v8f;
typedef __attribute__((ext_vector_type(4)))  unsigned u32x4;
typedef __attribute__((ext_vector_type(8)))  int      i32x8;
typedef __attribute__((ext_vector_type(4)))  int      i32x4;

// ---------------- helpers ----------------
__device__ __forceinline__ unsigned f2key(float f) {
    unsigned u = __float_as_uint(f);
    return (u & 0x80000000u) ? ~u : (u | 0x80000000u);   // monotone float->uint
}
__device__ __forceinline__ float key2f(unsigned k) {
    unsigned u = (k & 0x80000000u) ? (k & 0x7FFFFFFFu) : ~k;
    return __uint_as_float(u);
}
__device__ __forceinline__ void edge_sd(const int* __restrict__ ei, int e, int& s, int& d) {
    if (e < E_EDGES) { s = ei[e]; d = ei[E_EDGES + e]; }
    else             { s = e - E_EDGES; d = s; }         // self loop
}

// ---------------- elementwise utility kernels ----------------
__global__ void k_fill_u32(unsigned* __restrict__ p, unsigned v, int n) {
    int i = blockIdx.x * blockDim.x + threadIdx.x;
    if (i < n) p[i] = v;
}

__global__ void k_f32_to_f16(const float* __restrict__ in, _Float16* __restrict__ out, int n) {
    int i = blockIdx.x * blockDim.x + threadIdx.x;
    if (i < n) out[i] = (_Float16)in[i];
}

// ---------------- WMMA GEMM ----------------
// C[M_PAD, Ncols] = A[M_PAD, K](f16) x B[K, Ncols](f16) (+ bias[Ncols]).
// M is padded to a multiple of 512 rows so every wave's 4 tiles are valid:
// no divergent guards -> no accumulator phi copies, EXEC trivially all-1 for WMMA.
// Workgroup = 256 threads = 8 waves; each workgroup owns one 16-column strip of B.
// The K x 16 strip is DMA'd into LDS by the Tensor Data Mover (tensor_load_to_lds,
// TENSORcnt), then swizzled LDS->LDS into WMMA fragment order so each lane's B
// fragment is one contiguous 32-byte LDS read. Each wave computes FOUR 16x16
// M-tiles against the shared B fragment via v_wmma_f32_16x16x32_f16.
template <int K, bool WRITE16>
__global__ void __launch_bounds__(256)
k_wmma_gemm_f16(const _Float16* __restrict__ A,
                const _Float16* __restrict__ B,
                const float* __restrict__ bias,
                float* __restrict__ C,
                _Float16* __restrict__ C16,      // f16 mirror of C (if WRITE16)
                int Ncols) {
    __shared__ __align__(32) _Float16 ldsRaw[K * 16];        // linear K x 16 tile
    __shared__ __align__(32) _Float16 ldsB[(K / 32) * 512];  // fragment order

    const int nStrips = Ncols >> 4;
    const int nStrip  = blockIdx.x % nStrips;
    const int mPanel  = blockIdx.x / nStrips;

    // ---- stage B strip (K rows x 16 cols, row stride Ncols) into LDS ----
#if defined(__gfx1250__)
    if (threadIdx.x == 0) {
        // Tensor DMA descriptor (CDNA5 ISA 8.3/8.4): 2D tile, data_size=2B.
        const unsigned long long ga = (unsigned long long)(uintptr_t)(B + nStrip * 16);
        const unsigned lds_off = (unsigned)(uintptr_t)(&ldsRaw[0]);   // flat->LDS offset
        u32x4 g0;
        g0[0] = 1u;                                   // count=1, user descriptor
        g0[1] = lds_off;                              // lds_addr
        g0[2] = (unsigned)(ga & 0xFFFFFFFFu);         // global_addr[31:0]
        g0[3] = (unsigned)((ga >> 32) & 0x01FFFFFFu)  // global_addr[56:32]
              | (2u << 30);                           // type=2 ("image")
        i32x8 g1;
        g1[0] = (1 << 16);                            // data_size=1 -> 2 bytes
        g1[1] = (int)((16u & 0xFFFFu) << 16);         // tensor_dim0=16 (lo16)
        g1[2] = (int)(((unsigned)K & 0xFFFFu) << 16); // dim0 hi=0 | tensor_dim1=K (lo16)
        g1[3] = (int)(16u << 16);                     // dim1 hi=0 | tile_dim0=16
        g1[4] = (int)((unsigned)K & 0xFFFFu);         // tile_dim1=K | tile_dim2=0
        g1[5] = (int)(unsigned)Ncols;                 // tensor_dim0_stride lo32
        g1[6] = 0;                                    // stride0 hi | stride1 lo (unused, 2D)
        g1[7] = 0;
        i32x4 z4 = {0, 0, 0, 0};
        i32x8 z8 = {0, 0, 0, 0, 0, 0, 0, 0};
        __builtin_amdgcn_tensor_load_to_lds(g0, g1, z4, z4, z8, 0);  // 6-arg form
        __builtin_amdgcn_s_wait_tensorcnt(0);
    }
#else
    for (int idx = threadIdx.x; idx < K * 16; idx += 256)
        ldsRaw[idx] = B[(size_t)(idx >> 4) * Ncols + nStrip * 16 + (idx & 15)];
#endif
    __syncthreads();

    // ---- LDS->LDS swizzle into WMMA fragment order ----
    for (int idx = threadIdx.x; idx < K * 16; idx += 256) {
        int k = idx >> 4, col = idx & 15;
        int chunk = k >> 5, kk = k & 31;
        ldsB[chunk * 512 + (col + ((kk & 8) << 1)) * 16 + (kk & 7) + ((kk & 16) >> 1)] = ldsRaw[idx];
    }
    __syncthreads();

    const int waveId = threadIdx.x >> 5;
    const int lane   = threadIdx.x & 31;
    const int hw     = lane >> 4;
    const int l16    = lane & 15;
    const int tile0  = mPanel * 32 + waveId * 4;  // all 4 tiles valid (M padded)

    v8f cacc[4] = {{}, {}, {}, {}};
#pragma unroll
    for (int kb = 0; kb < K; kb += 32) {
        // shared B fragment: one contiguous 32B LDS read per lane
        v16h b = *(const v16h*)&ldsB[(kb >> 5) * 512 + lane * 16];
#pragma unroll
        for (int j = 0; j < 4; ++j) {
            const _Float16* ap = A + (size_t)((tile0 + j) * 16 + l16) * K + kb + hw * 8;
            if (kb + 32 < K)                 // prefetch next K chunk (global_prefetch_b8)
                __builtin_prefetch((const void*)(ap + 32), 0, 3);
            v8h alo = *(const v8h*)(ap);
            v8h ahi = *(const v8h*)(ap + 16);
            v16h a;
#pragma unroll
            for (int i = 0; i < 8; ++i) { a[i] = alo[i]; a[8 + i] = ahi[i]; }
            cacc[j] = __builtin_amdgcn_wmma_f32_16x16x32_f16(false, a, false, b,
                                                             (short)0, cacc[j], false, false);
        }
    }

    // ---- store 4 tiles (branch-free) ----
    const int   bcol = nStrip * 16 + l16;
    const float bv   = bias ? bias[bcol] : 0.0f;
#pragma unroll
    for (int j = 0; j < 4; ++j) {
        const int crow0 = (tile0 + j) * 16 + hw * 8;
#pragma unroll
        for (int r = 0; r < 8; ++r) {
            float val = cacc[j][r] + bv;
            size_t idx = (size_t)(crow0 + r) * Ncols + bcol;
            C[idx] = val;
            if (WRITE16) C16[idx] = (_Float16)val;
        }
    }
}

// ---------------- attention coefficients ----------------
// One wave per (node, head): alpha_s = <hp[n,h,:], a_src[h,:]>, alpha_d likewise.
__global__ void k_alpha(const float* __restrict__ hp,
                        const float* __restrict__ asrc,
                        const float* __restrict__ adst,
                        float* __restrict__ out_s,
                        float* __restrict__ out_d) {
    int wave = blockIdx.x * (blockDim.x >> 5) + (threadIdx.x >> 5);
    int node = wave >> 2;
    int head = wave & 3;
    if (node >= N_NODES) return;
    int lane = threadIdx.x & 31;

    const float* hrow = hp + (size_t)node * (HEADS * HID) + head * HID;
    const float* s_w  = asrc + head * HID;
    const float* d_w  = adst + head * HID;
    float ss = 0.f, sd = 0.f;
#pragma unroll
    for (int i = 0; i < 4; ++i) {
        float v = hrow[lane + 32 * i];
        ss += v * s_w[lane + 32 * i];
        sd += v * d_w[lane + 32 * i];
    }
#pragma unroll
    for (int off = 16; off; off >>= 1) {
        ss += __shfl_xor(ss, off, 32);
        sd += __shfl_xor(sd, off, 32);
    }
    if (lane == 0) {
        out_s[node * HEADS + head] = ss;
        out_d[node * HEADS + head] = sd;
    }
}

// ---------------- segment softmax: max pass ----------------
__global__ void k_edge_max(const int* __restrict__ ei,
                           const float* __restrict__ as, const float* __restrict__ ad,
                           unsigned* __restrict__ maxk) {
    int i = blockIdx.x * blockDim.x + threadIdx.x;
    if (i >= E_TOT * HEADS) return;
    int e = i >> 2, h = i & 3;
    int s, d; edge_sd(ei, e, s, d);
    float ev = as[s * HEADS + h] + ad[d * HEADS + h];
    ev = ev > 0.f ? ev : ev * NEG_SLOPE;
    atomicMax(&maxk[d * HEADS + h], f2key(ev));
}

// ---------------- segment softmax: exp + denom pass ----------------
__global__ void k_edge_expsum(const int* __restrict__ ei,
                              const float* __restrict__ as, const float* __restrict__ ad,
                              const unsigned* __restrict__ maxk,
                              float* __restrict__ eexp, float* __restrict__ denom) {
    int i = blockIdx.x * blockDim.x + threadIdx.x;
    if (i >= E_TOT * HEADS) return;
    int e = i >> 2, h = i & 3;
    int s, d; edge_sd(ei, e, s, d);
    float ev = as[s * HEADS + h] + ad[d * HEADS + h];
    ev = ev > 0.f ? ev : ev * NEG_SLOPE;
    float m  = key2f(maxk[d * HEADS + h]);
    float ex = __expf(ev - m);
    eexp[(size_t)e * HEADS + h] = ex;
    atomicAdd(&denom[d * HEADS + h], ex);
}

// ---------------- weighted message scatter (one wave per edge) ----------------
__global__ void k_edge_scatter(const int* __restrict__ ei,
                               const float* __restrict__ hp,
                               const float* __restrict__ eexp,
                               const float* __restrict__ denom,
                               float* __restrict__ acc) {
    int wave = blockIdx.x * (blockDim.x >> 5) + (threadIdx.x >> 5);
    if (wave >= E_TOT) return;
    int lane = threadIdx.x & 31;
    int s, d; edge_sd(ei, wave, s, d);

    float a0 = eexp[(size_t)wave * HEADS + 0] / (denom[d * HEADS + 0] + 1e-16f);
    float a1 = eexp[(size_t)wave * HEADS + 1] / (denom[d * HEADS + 1] + 1e-16f);
    float a2 = eexp[(size_t)wave * HEADS + 2] / (denom[d * HEADS + 2] + 1e-16f);
    float a3 = eexp[(size_t)wave * HEADS + 3] / (denom[d * HEADS + 3] + 1e-16f);

    const float* hs = hp + (size_t)s * (HEADS * HID);
    float* ac = acc + (size_t)d * HID;
#pragma unroll
    for (int i = 0; i < 4; ++i) {
        int c = lane + 32 * i;
        float v = a0 * hs[c] + a1 * hs[HID + c] + a2 * hs[2 * HID + c] + a3 * hs[3 * HID + c];
        atomicAdd(&ac[c], v);
    }
}

// ---------------- head-mean + bias + relu + residual + layernorm ----------------
// Also emits the f16 mirror of h consumed by the next layer's WMMA GEMM.
__global__ void k_post_ln(const float* __restrict__ acc,
                          const float* __restrict__ cb,
                          const float* __restrict__ g, const float* __restrict__ b,
                          float* __restrict__ h, _Float16* __restrict__ h16) {
    __shared__ float red[HID];
    int n = blockIdx.x, t = threadIdx.x;
    float v = acc[(size_t)n * HID + t] * (1.0f / HEADS) + cb[t];
    v = fmaxf(v, 0.f) + h[(size_t)n * HID + t];

    red[t] = v; __syncthreads();
    for (int s = HID / 2; s > 0; s >>= 1) { if (t < s) red[t] += red[t + s]; __syncthreads(); }
    float mu = red[0] * (1.0f / HID);
    __syncthreads();
    float dv = v - mu;
    red[t] = dv * dv; __syncthreads();
    for (int s = HID / 2; s > 0; s >>= 1) { if (t < s) red[t] += red[t + s]; __syncthreads(); }
    float var = red[0] * (1.0f / HID);
    float outv = dv * rsqrtf(var + LN_EPS) * g[t] + b[t];
    h[(size_t)n * HID + t]   = outv;
    h16[(size_t)n * HID + t] = (_Float16)outv;
}

// ---------------- global mean pool ----------------
__global__ void k_pool(const float* __restrict__ h, const int* __restrict__ batch,
                       float* __restrict__ sums, float* __restrict__ cnt) {
    int n = blockIdx.x, t = threadIdx.x;
    int gidx = batch[n];
    atomicAdd(&sums[(size_t)gidx * HID + t], h[(size_t)n * HID + t]);
    if (t == 0) atomicAdd(&cnt[gidx], 1.0f);
}

__global__ void k_pool_div(const float* __restrict__ sums, const float* __restrict__ cnt,
                           float* __restrict__ out) {
    int i = blockIdx.x * blockDim.x + threadIdx.x;
    if (i >= N_GRAPHS * HID) return;
    out[i] = sums[i] / fmaxf(cnt[i >> 7], 1.0f);
}

// ---------------- host launch ----------------
extern "C" void kernel_launch(void* const* d_in, const int* in_sizes, int n_in,
                              void* d_out, int out_size, void* d_ws, size_t ws_size,
                              hipStream_t stream) {
    (void)in_sizes; (void)n_in; (void)out_size; (void)ws_size;

    const float* x        = (const float*)d_in[0];
    const int*   edge_idx = (const int*)d_in[1];   // [2, E] (per harness: integer -> int*)
    const int*   batch    = (const int*)d_in[2];   // [N]
    const float* W_in     = (const float*)d_in[3]; // [64,128]
    const float* b_in     = (const float*)d_in[4]; // [128]
    const float* Ws       = (const float*)d_in[5]; // [3,128,512]
    const float* att_src  = (const float*)d_in[6]; // [3,4,128]
    const float* att_dst  = (const float*)d_in[7]; // [3,4,128]
    const float* conv_b   = (const float*)d_in[8]; // [3,128]
    const float* ln_g     = (const float*)d_in[9]; // [3,128]
    const float* ln_b     = (const float*)d_in[10];// [3,128]
    float*       out      = (float*)d_out;         // [64,128]

    // ---- workspace carving (256B aligned); M-dim buffers padded to M_PAD ----
    char* w = (char*)d_ws;
    size_t off = 0;
    auto take = [&](size_t bytes) { size_t o = off; off = (off + bytes + 255) & ~(size_t)255; return o; };

    float*     h      = (float*)(w + take((size_t)M_PAD * HID * 4));
    float*     hp     = (float*)(w + take((size_t)M_PAD * HEADS * HID * 4));
    _Float16*  hf16   = (_Float16*)(w + take((size_t)M_PAD * HID * 2));
    _Float16*  xf16   = (_Float16*)(w + take((size_t)M_PAD * IN_DIM * 2));
    _Float16*  winf16 = (_Float16*)(w + take((size_t)IN_DIM * HID * 2));
    _Float16*  wsf16  = (_Float16*)(w + take((size_t)LAYERS * HID * HEADS * HID * 2));
    float*     a_s    = (float*)(w + take((size_t)N_NODES * HEADS * 4));
    float*     a_d    = (float*)(w + take((size_t)N_NODES * HEADS * 4));
    unsigned*  maxk   = (unsigned*)(w + take((size_t)N_NODES * HEADS * 4));
    float*     denom  = (float*)(w + take((size_t)N_NODES * HEADS * 4));
    float*     eexp   = (float*)(w + take((size_t)E_TOT * HEADS * 4));
    float*     acc    = (float*)(w + take((size_t)N_NODES * HID * 4));
    float*     sums   = (float*)(w + take((size_t)N_GRAPHS * HID * 4));
    float*     cnt    = (float*)(w + take((size_t)N_GRAPHS * 4));

    auto cdiv = [](int a, int b) { return (a + b - 1) / b; };

    const int M_PANELS = M_PAD / 512;        // 40 panels (8 waves x 4 tiles x 16 rows)

    // ---- f16 weight/activation staging (weights tiny; x once) ----
    k_f32_to_f16<<<cdiv(IN_DIM * HID, 256), 256, 0, stream>>>(W_in, winf16, IN_DIM * HID);
    k_f32_to_f16<<<cdiv(LAYERS * HID * HEADS * HID, 256), 256, 0, stream>>>(Ws, wsf16, LAYERS * HID * HEADS * HID);
    k_f32_to_f16<<<cdiv(N_NODES * IN_DIM, 256), 256, 0, stream>>>(x, xf16, N_NODES * IN_DIM);

    // ---- input projection: h = x @ W_in + b_in (K=64, N=128); also emits hf16 ----
    k_wmma_gemm_f16<IN_DIM, true><<<(HID / 16) * M_PANELS, 256, 0, stream>>>(
        xf16, winf16, b_in, h, hf16, HID);

    // ---- GAT layers ----
    for (int l = 0; l < LAYERS; ++l) {
        // hp = h @ Ws[l]  (K=128, N=512)
        k_wmma_gemm_f16<HID, false><<<((HEADS * HID) / 16) * M_PANELS, 256, 0, stream>>>(
            hf16, wsf16 + (size_t)l * HID * HEADS * HID, nullptr, hp, nullptr, HEADS * HID);

        // attention logits per (node, head)
        k_alpha<<<cdiv(N_NODES * HEADS, 8), 256, 0, stream>>>(
            hp, att_src + l * HEADS * HID, att_dst + l * HEADS * HID, a_s, a_d);

        // init segment-softmax state
        k_fill_u32<<<cdiv(N_NODES * HEADS, 256), 256, 0, stream>>>(maxk, 0x007FFFFFu, N_NODES * HEADS); // key(-inf)
        k_fill_u32<<<cdiv(N_NODES * HEADS, 256), 256, 0, stream>>>((unsigned*)denom, 0u, N_NODES * HEADS);
        k_fill_u32<<<cdiv(N_NODES * HID, 256), 256, 0, stream>>>((unsigned*)acc, 0u, N_NODES * HID);

        k_edge_max<<<cdiv(E_TOT * HEADS, 256), 256, 0, stream>>>(edge_idx, a_s, a_d, maxk);
        k_edge_expsum<<<cdiv(E_TOT * HEADS, 256), 256, 0, stream>>>(edge_idx, a_s, a_d, maxk, eexp, denom);
        k_edge_scatter<<<cdiv(E_TOT, 8), 256, 0, stream>>>(edge_idx, hp, eexp, denom, acc);

        // head-mean + relu + residual + LN; also refresh hf16 for next layer's GEMM
        k_post_ln<<<N_NODES, HID, 0, stream>>>(
            acc, conv_b + l * HID, ln_g + l * HID, ln_b + l * HID, h, hf16);
    }

    // ---- global mean pool ----
    k_fill_u32<<<cdiv(N_GRAPHS * HID, 256), 256, 0, stream>>>((unsigned*)sums, 0u, N_GRAPHS * HID);
    k_fill_u32<<<1, 256, 0, stream>>>((unsigned*)cnt, 0u, N_GRAPHS);
    k_pool<<<N_NODES, HID, 0, stream>>>(h, batch, sums, cnt);
    k_pool_div<<<cdiv(N_GRAPHS * HID, 256), 256, 0, stream>>>(sums, cnt, out);
}
